// LSTMDecoder_24575802868292
// MI455X (gfx1250) — matrix operand order repaired and checked by reading, not compiled, not run on hardware
//
#include <hip/hip_runtime.h>
#include <hip/hip_bf16.h>
#include <math.h>

typedef __attribute__((ext_vector_type(16))) __bf16 v16bf;
typedef __attribute__((ext_vector_type(8)))  __bf16 v8bf;
typedef __attribute__((ext_vector_type(8)))  float  v8f;

#define B_     64
#define S_     64
#define T_     21
#define D_     512          // ENC_D = HID = EMB
#define STYLE_ 128
#define V_     32000
#define KCAT   1536         // [context(512) | emb(512) | h(512)]
#define GATES_ 2048
#define AP_G   1544         // padded LDS row stride (bf16 elems), rows stay 16B-aligned
#define AP_P   520

// ---------------------------------------------------------------- utilities
__device__ __forceinline__ float sigmoidf_(float x) { return 1.f / (1.f + __expf(-x)); }

// async global->LDS 16B chunk (CDNA5 GLOBAL_LOAD_ASYNC_TO_LDS_B128, GVS mode)
__device__ __forceinline__ void async_cp16(unsigned lds_off, unsigned gbl_off,
                                           unsigned long long base) {
  asm volatile("global_load_async_to_lds_b128 %0, %1, %2"
               :: "v"(lds_off), "v"(gbl_off), "s"(base) : "memory");
}
__device__ __forceinline__ void async_wait0() {
  asm volatile("s_wait_asynccnt 0" ::: "memory");
}

__global__ void k_zero(float* p, int n) {
  int i = blockIdx.x * blockDim.x + threadIdx.x;
  if (i < n) p[i] = 0.f;
}

__global__ void k_cvt_bf16(const float* __restrict__ src, __bf16* __restrict__ dst, size_t n) {
  size_t i = (size_t)blockIdx.x * blockDim.x + threadIdx.x;
  size_t stride = (size_t)gridDim.x * blockDim.x;
  for (; i < n; i += stride) dst[i] = (__bf16)src[i];
}

// wcat[g][k] = k<1024 ? w_ih[g][k] : w_hh[g][k-1024]   (row-major, bf16)
__global__ void k_wcat(const float* __restrict__ wih, const float* __restrict__ whh,
                       __bf16* __restrict__ wcat) {
  size_t i = (size_t)blockIdx.x * blockDim.x + threadIdx.x;
  size_t stride = (size_t)gridDim.x * blockDim.x;
  size_t n = (size_t)GATES_ * KCAT;
  for (; i < n; i += stride) {
    int g = (int)(i / KCAT), k = (int)(i % KCAT);
    float v = (k < 1024) ? wih[(size_t)g * 1024 + k] : whh[(size_t)g * 512 + (k - 1024)];
    wcat[i] = (__bf16)v;
  }
}

// h0 = [encode_hidden | style_emb[:,3,:]] @ W_e2d_w.T + b ; c0 = 0 ; seed x_bf h-slot
__global__ void k_h0(const float* __restrict__ eh, const float* __restrict__ st,
                     const float* __restrict__ W, const float* __restrict__ bias,
                     float* __restrict__ h, float* __restrict__ c, __bf16* __restrict__ xbf) {
  int idx = blockIdx.x * blockDim.x + threadIdx.x;   // B_*D_ threads
  if (idx >= B_ * D_) return;
  int b = idx >> 9, j = idx & 511;
  const float* wr = W + (size_t)j * (D_ + STYLE_);
  const float* er = eh + (size_t)b * D_;
  const float* sr = st + ((size_t)b * 4 + 3) * STYLE_;
  float acc = bias[j];
  for (int k = 0; k < D_; ++k)      acc += er[k] * wr[k];
  for (int k = 0; k < STYLE_; ++k)  acc += sr[k] * wr[D_ + k];
  h[idx] = acc;
  c[idx] = 0.f;
  xbf[(size_t)b * KCAT + 1024 + j] = (__bf16)acc;
}

// emb_bf[b,t,:] = bf16(emb_table[decoder_input[b,t]])
__global__ void k_emb(const int* __restrict__ dec, const float* __restrict__ tab,
                      __bf16* __restrict__ embbf) {
  int i = blockIdx.x * blockDim.x + threadIdx.x;     // B_*T_*D_ threads
  if (i >= B_ * T_ * D_) return;
  int d = i & 511, bt = i >> 9;
  embbf[i] = (__bf16)tab[(size_t)dec[bt] * D_ + d];
}

// hW[b,j] = sum_i h[b,i] * attn_W[i,j]
__global__ void k_hW(const float* __restrict__ h, const float* __restrict__ aW,
                     float* __restrict__ hW) {
  int b = blockIdx.x, j = threadIdx.x;               // <<<64,512>>>
  const float* hr = h + (size_t)b * D_;
  float acc = 0.f;
  for (int i = 0; i < D_; ++i) acc += hr[i] * aW[(size_t)i * D_ + j];
  hW[(size_t)b * D_ + j] = acc;
}

// per-b: scores -> softmax -> context ; write x_bf[b] = [bf16(ctx) | emb_bf[b,t]]
__global__ void k_attn(const float* __restrict__ hW, const float* __restrict__ eo,
                       const __bf16* __restrict__ embbf, __bf16* __restrict__ xbf, int t) {
  int b = blockIdx.x, tid = threadIdx.x;             // <<<64,256>>>
  __shared__ float hwl[D_];
  __shared__ float sc[S_];
  for (int k = tid; k < D_; k += 256) hwl[k] = hW[(size_t)b * D_ + k];
  __syncthreads();
  int s = tid >> 2, part = tid & 3;                  // 4 threads per s
  const float* er = eo + ((size_t)b * S_ + s) * D_;
  float acc = 0.f;
  for (int k = part; k < D_; k += 4) acc += er[k] * hwl[k];
  acc += __shfl_xor(acc, 1);
  acc += __shfl_xor(acc, 2);
  if (part == 0) sc[s] = acc;
  __syncthreads();
  if (tid == 0) {
    float m = -3.4e38f;
    for (int i = 0; i < S_; ++i) m = fmaxf(m, sc[i]);
    float ss = 0.f;
    for (int i = 0; i < S_; ++i) ss += __expf(sc[i] - m);
    float inv = 1.f / ss;
    for (int i = 0; i < S_; ++i) sc[i] = __expf(sc[i] - m) * inv;
  }
  __syncthreads();
  for (int d = tid; d < D_; d += 256) {
    float cx = 0.f;
    for (int s2 = 0; s2 < S_; ++s2) cx += sc[s2] * eo[((size_t)b * S_ + s2) * D_ + d];
    xbf[(size_t)b * KCAT + d]       = (__bf16)cx;
    xbf[(size_t)b * KCAT + D_ + d]  = embbf[((size_t)b * T_ + t) * D_ + d];
  }
}

// gates(64x2048) = X(64x1536) @ Wcat^T + b_ih + b_hh, bf16 WMMA
__global__ void __launch_bounds__(256) k_gates(const __bf16* __restrict__ xbf,
                                               const __bf16* __restrict__ wcat,
                                               const float* __restrict__ bih,
                                               const float* __restrict__ bhh,
                                               float* __restrict__ gates) {
  __shared__ __bf16 A[16 * AP_G];
  int mi = blockIdx.x & 3, ni = blockIdx.x >> 2;     // <<<64,256>>>
  int tid = threadIdx.x, wave = tid >> 5, lane = tid & 31;
  int row0 = mi * 16;
  // async stage A tile: 16 rows x 1536 bf16 = 48KB = 3072 x 16B chunks
  unsigned long long xbase = (unsigned long long)(uintptr_t)xbf;
  for (int i = 0; i < 12; ++i) {
    int idx = tid + 256 * i;                         // chunk id
    int r = idx / 192, k = (idx % 192) << 3;         // 192 chunks of 8 bf16 per row
    unsigned ldso = (unsigned)(uintptr_t)&A[r * AP_G + k];
    unsigned gblo = ((unsigned)(row0 + r) * KCAT + (unsigned)k) * 2u;
    async_cp16(ldso, gblo, xbase);
  }
  async_wait0();
  __syncthreads();
  int m    = lane & 15;
  int koff = (lane < 16) ? 0 : 16;                   // A: K half per lane group
  int kb   = (lane < 16) ? 0 : 16;                   // B: K half per lane group
  int n    = ni * 128 + wave * 16 + (lane & 15);
  v8f c = {};
  for (int k0 = 0; k0 < KCAT; k0 += 32) {
    v8bf lo = *(const v8bf*)&A[m * AP_G + k0 + koff];
    v8bf hi = *(const v8bf*)&A[m * AP_G + k0 + 8 + koff];
    v16bf a;
    for (int j = 0; j < 8; ++j) { a[j] = lo[j]; a[8 + j] = hi[j]; }
    v16bf bm = *(const v16bf*)(wcat + (size_t)n * KCAT + k0 + kb);
    c = __builtin_amdgcn_wmma_f32_16x16x32_bf16(false, a, false, bm, (short)0, c, false, false);
  }
  float bias = bih[n] + bhh[n];
  int rbase = (lane < 16) ? 0 : 8;
  for (int r = 0; r < 8; ++r)
    gates[(size_t)(row0 + rbase + r) * GATES_ + n] = c[r] + bias;
}

// LSTM cell elementwise; writes h, c, outs_bf, and x_bf h-slot for next step
__global__ void k_cell(const float* __restrict__ g, float* __restrict__ c,
                       float* __restrict__ h, __bf16* __restrict__ outsbf,
                       __bf16* __restrict__ xbf, int t) {
  int idx = blockIdx.x * blockDim.x + threadIdx.x;   // B_*D_ threads
  if (idx >= B_ * D_) return;
  int b = idx >> 9, j = idx & 511;
  const float* gr = g + (size_t)b * GATES_;
  float gi = gr[j], gf = gr[D_ + j], gg = gr[2 * D_ + j], go = gr[3 * D_ + j];
  float cn = sigmoidf_(gf) * c[idx] + sigmoidf_(gi) * tanhf(gg);
  float hn = sigmoidf_(go) * tanhf(cn);
  c[idx] = cn;
  h[idx] = hn;
  outsbf[((size_t)b * T_ + t) * D_ + j] = (__bf16)hn;
  xbf[(size_t)b * KCAT + 1024 + j]      = (__bf16)hn;
}

// fused projection GEMM (bf16 WMMA, A hoisted to VGPRs) + per-lane online logsumexp
// + exact label logit + masked NLL
__global__ void __launch_bounds__(256) k_proj_loss(const __bf16* __restrict__ outsbf,
                                                   const __bf16* __restrict__ projbf,
                                                   const float* __restrict__ pb,
                                                   const float* __restrict__ projw,
                                                   const int* __restrict__ label,
                                                   float* __restrict__ nllbuf) {
  __shared__ __bf16 A[16 * AP_P];
  __shared__ float redm[8][16];
  __shared__ float reds[8][16];
  __shared__ float Mfin[16];
  __shared__ float Sfin[16];
  int tid = threadIdx.x, wave = tid >> 5, lane = tid & 31;
  int row0 = blockIdx.x * 16;                        // <<<84,256>>>, 84*16 == 1344
  // async stage A tile: 16 rows x 512 bf16 = 16KB = 1024 x 16B chunks
  unsigned long long obase = (unsigned long long)(uintptr_t)outsbf;
  for (int i = 0; i < 4; ++i) {
    int idx = tid + 256 * i;
    int r = idx >> 6, k = (idx & 63) << 3;           // 64 chunks of 8 bf16 per row
    unsigned ldso = (unsigned)(uintptr_t)&A[r * AP_P + k];
    unsigned gblo = ((unsigned)(row0 + r) * D_ + (unsigned)k) * 2u;
    async_cp16(ldso, gblo, obase);
  }
  async_wait0();
  __syncthreads();

  int m     = lane & 15;
  int koff  = (lane < 16) ? 0 : 16;
  int kb    = (lane < 16) ? 0 : 16;
  int rbase = (lane < 16) ? 0 : 8;

  // hoist all 16 A fragments into VGPRs (constant across the whole vocab loop)
  v16bf areg[16];
#pragma unroll
  for (int kk = 0; kk < 16; ++kk) {
    int k0 = kk * 32;
    v8bf lo = *(const v8bf*)&A[m * AP_P + k0 + koff];
    v8bf hi = *(const v8bf*)&A[m * AP_P + k0 + 8 + koff];
    for (int j = 0; j < 8; ++j) { areg[kk][j] = lo[j]; areg[kk][8 + j] = hi[j]; }
  }

  // per-lane running logsumexp state (lanes see disjoint vocab columns)
  float mrun[8], srun[8];
#pragma unroll
  for (int r = 0; r < 8; ++r) { mrun[r] = -3.4e38f; srun[r] = 0.f; }

  for (int n0 = 0; n0 < V_; n0 += 128) {
    int n = n0 + wave * 16 + (lane & 15);
    if (n0 + 128 < V_ && tid < 128)                  // hint next 128 proj rows into L2
      __builtin_prefetch(projbf + (size_t)(n0 + 128 + tid) * D_, 0, 1);
    v8f c = {};
#pragma unroll
    for (int kk = 0; kk < 16; ++kk) {
      v16bf bm = *(const v16bf*)(projbf + (size_t)n * D_ + kk * 32 + kb);
      c = __builtin_amdgcn_wmma_f32_16x16x32_bf16(false, areg[kk], false, bm,
                                                  (short)0, c, false, false);
    }
    float bias = pb[n];
#pragma unroll
    for (int r = 0; r < 8; ++r) {
      float v  = c[r] + bias;
      float nm = fmaxf(mrun[r], v);
      srun[r]  = srun[r] * __expf(mrun[r] - nm) + __expf(v - nm);
      mrun[r]  = nm;
    }
  }

  // combine (m,s) across the 16 lanes of each half-wave
#pragma unroll
  for (int mask = 1; mask <= 8; mask <<= 1) {
#pragma unroll
    for (int r = 0; r < 8; ++r) {
      float om = __shfl_xor(mrun[r], mask);
      float os = __shfl_xor(srun[r], mask);
      float nm = fmaxf(mrun[r], om);
      srun[r]  = srun[r] * __expf(mrun[r] - nm) + os * __expf(om - nm);
      mrun[r]  = nm;
    }
  }
  if ((lane & 15) == 0) {
#pragma unroll
    for (int r = 0; r < 8; ++r) { redm[wave][rbase + r] = mrun[r]; reds[wave][rbase + r] = srun[r]; }
  }
  __syncthreads();
  if (tid < 16) {                                    // combine the 8 waves
    float M = redm[0][tid], Ssum = reds[0][tid];
    for (int w = 1; w < 8; ++w) {
      float om = redm[w][tid], os = reds[w][tid];
      float nm = fmaxf(M, om);
      Ssum = Ssum * __expf(M - nm) + os * __expf(om - nm);
      M = nm;
    }
    Mfin[tid] = M; Sfin[tid] = Ssum;
  }
  __syncthreads();

  // exact label logit (fp32 proj_w row) + masked NLL accumulation
  int mr = tid >> 4, ln = tid & 15;                  // 16 threads per row
  int row = row0 + mr;
  int lbl = label[row];
  float acc = 0.f;
  const float* wr = projw + (size_t)lbl * D_;
  for (int k = ln; k < D_; k += 16) acc += (float)A[mr * AP_P + k] * wr[k];
  acc += __shfl_xor(acc, 1); acc += __shfl_xor(acc, 2);
  acc += __shfl_xor(acc, 4); acc += __shfl_xor(acc, 8);
  if (ln == 0) {
    float logit = acc + pb[lbl];
    float nll   = (Mfin[mr] + __logf(Sfin[mr])) - logit;
    float maskv = (lbl > 0) ? 1.f : 0.f;
    int b = row / T_;
    atomicAdd(&nllbuf[b], maskv * nll);
    atomicAdd(&nllbuf[B_ + b], maskv);
  }
}

__global__ void k_final(const float* __restrict__ nllbuf, float* __restrict__ out) {
  __shared__ float red[B_];
  int b = threadIdx.x;                               // <<<1,64>>>
  red[b] = nllbuf[b] / (nllbuf[B_ + b] + 1e-6f);
  __syncthreads();
  if (b == 0) {
    float s = 0.f;
    for (int i = 0; i < B_; ++i) s += red[i];
    out[0] = s / (float)B_;
  }
}

// ---------------------------------------------------------------- launch
extern "C" void kernel_launch(void* const* d_in, const int* in_sizes, int n_in,
                              void* d_out, int out_size, void* d_ws, size_t ws_size,
                              hipStream_t stream) {
  const float* enc_h   = (const float*)d_in[0];
  const float* enc_out = (const float*)d_in[1];
  const int*   seq_lab = (const int*)  d_in[3];
  const int*   dec_in  = (const int*)  d_in[4];
  const float* style   = (const float*)d_in[5];
  const float* W_e2d   = (const float*)d_in[6];
  const float* b_e2d   = (const float*)d_in[7];
  const float* attnW   = (const float*)d_in[8];
  const float* embtab  = (const float*)d_in[9];
  const float* w_ih    = (const float*)d_in[10];
  const float* w_hh    = (const float*)d_in[11];
  const float* b_ih    = (const float*)d_in[12];
  const float* b_hh    = (const float*)d_in[13];
  const float* proj_w  = (const float*)d_in[14];
  const float* proj_b  = (const float*)d_in[15];
  float* out = (float*)d_out;

  char* p = (char*)d_ws;
  auto alloc = [&](size_t bytes) { char* r = p; p += (bytes + 255) & ~(size_t)255; return (void*)r; };
  float*  h      = (float*) alloc((size_t)B_ * D_ * 4);
  float*  c      = (float*) alloc((size_t)B_ * D_ * 4);
  float*  hW     = (float*) alloc((size_t)B_ * D_ * 4);
  float*  gates  = (float*) alloc((size_t)B_ * GATES_ * 4);
  float*  nllbuf = (float*) alloc((size_t)2 * B_ * 4);
  __bf16* xbf    = (__bf16*)alloc((size_t)B_ * KCAT * 2);
  __bf16* embbf  = (__bf16*)alloc((size_t)B_ * T_ * D_ * 2);
  __bf16* outsbf = (__bf16*)alloc((size_t)B_ * T_ * D_ * 2);
  __bf16* wcat   = (__bf16*)alloc((size_t)GATES_ * KCAT * 2);
  __bf16* projbf = (__bf16*)alloc((size_t)V_ * D_ * 2);

  k_zero<<<1, 128, 0, stream>>>(nllbuf, 2 * B_);
  k_wcat<<<3072, 256, 0, stream>>>(w_ih, w_hh, wcat);
  k_cvt_bf16<<<4096, 256, 0, stream>>>(proj_w, projbf, (size_t)V_ * D_);
  k_h0<<<(B_ * D_ + 255) / 256, 256, 0, stream>>>(enc_h, style, W_e2d, b_e2d, h, c, xbf);
  k_emb<<<(B_ * T_ * D_ + 255) / 256, 256, 0, stream>>>(dec_in, embtab, embbf);

  for (int t = 0; t < T_; ++t) {
    k_hW  <<<B_, 512, 0, stream>>>(h, attnW, hW);
    k_attn<<<B_, 256, 0, stream>>>(hW, enc_out, embbf, xbf, t);
    k_gates<<<64, 256, 0, stream>>>(xbf, wcat, b_ih, b_hh, gates);
    k_cell<<<(B_ * D_ + 255) / 256, 256, 0, stream>>>(gates, c, h, outsbf, xbf, t);
  }

  k_proj_loss<<<(B_ * T_) / 16, 256, 0, stream>>>(outsbf, projbf, proj_b, proj_w, seq_lab, nllbuf);
  k_final<<<1, 64, 0, stream>>>(nllbuf, out);
}
